// AnalysisConvolutionBlock_26852135535105
// MI455X (gfx1250) — matrix operand order repaired and checked by reading, not compiled, not run on hardware
//
#include <hip/hip_runtime.h>
#include <hip/hip_bf16.h>

// ---------------------------------------------------------------------------
// Types for CDNA5 WMMA
// ---------------------------------------------------------------------------
typedef _Float16 v16h __attribute__((ext_vector_type(16)));
typedef _Float16 v8h  __attribute__((ext_vector_type(8)));
typedef float    v8f  __attribute__((ext_vector_type(8)));

// ---------------------------------------------------------------------------
// Problem constants (from reference setup_inputs)
// ---------------------------------------------------------------------------
#define B_   8
#define CIN  64
#define COUT 128
#define H_   256
#define W_   256
#define KTOT 576          // CIN * 9
#define HH   128          // downsampled
#define WH   128

// d_out layout (concatenated tuple, float32)
#define SZ_SUB  16777216  // 8*128*128*128
#define SZ_MSK  131072    // 8*1*128*128
#define OFF_LL  0
#define OFF_MLL 16777216
#define OFF_LH  16908288
#define OFF_HL  33685504
#define OFF_HH  50462720
#define OFF_MLH 67239936
#define OFF_MHL 67371008
#define OFF_MHH 67502080

// CDF 9/7 analysis filters
__device__ __constant__ float cH0[9] = {
    0.026748757410810f, -0.016864118442875f, -0.078223266528990f,
    0.266864118442875f,  0.602949018236360f,  0.266864118442875f,
   -0.078223266528990f, -0.016864118442875f,  0.026748757410810f};
__device__ __constant__ float cH1[7] = {
    0.091271763114250f, -0.057543526228500f, -0.591271763114250f,
    1.115087052457000f, -0.591271763114250f, -0.057543526228500f,
    0.091271763114250f};

__device__ __forceinline__ int refl(int i, int n) {
    if (i < 0) i = -i;
    if (i >= n) i = 2 * n - 2 - i;
    return i;
}

// ---------------------------------------------------------------------------
// K0: spectral norm (one power-iteration step with fixed u) -> 1/sigma
// ---------------------------------------------------------------------------
__global__ __launch_bounds__(256) void sigma_kernel(const float* __restrict__ w,
                                                    const float* __restrict__ u,
                                                    float* __restrict__ out) {
    __shared__ float vsh[KTOT];
    __shared__ float red[256];
    const int tid = threadIdx.x;
    for (int k = tid; k < KTOT; k += 256) {
        float s = 0.f;
        for (int o = 0; o < COUT; ++o) s += w[o * KTOT + k] * u[o];
        vsh[k] = s;
    }
    __syncthreads();
    float loc = 0.f;
    for (int k = tid; k < KTOT; k += 256) loc += vsh[k] * vsh[k];
    red[tid] = loc;
    __syncthreads();
    for (int s = 128; s > 0; s >>= 1) {
        if (tid < s) red[tid] += red[tid + s];
        __syncthreads();
    }
    const float vnorm = sqrtf(red[0]) + 1e-12f;
    __syncthreads();
    float loc2 = 0.f;
    for (int o = tid; o < COUT; o += 256) {
        float d = 0.f;
        for (int k = 0; k < KTOT; ++k) d += w[o * KTOT + k] * vsh[k];
        loc2 += d * d;
    }
    red[tid] = loc2;
    __syncthreads();
    for (int s = 128; s > 0; s >>= 1) {
        if (tid < s) red[tid] += red[tid + s];
        __syncthreads();
    }
    if (tid == 0) out[0] = vnorm / sqrtf(red[0]);  // 1/sigma
}

// ---------------------------------------------------------------------------
// K1: pack conv weight [128,576] fp32 -> f16 WMMA B fragments.
// B 32x16 f16 layout: lanes 0-15 -> n=lane, K=K0..K0+7, K0+16..K0+23 (pairs per VGPR);
// lanes 16-31 -> n=lane-16, K=K0+8..K0+15, K0+24..K0+31.
// packW[((kt*8+nt)*32+lane)*16 + 2j (+1)]
// ---------------------------------------------------------------------------
__global__ __launch_bounds__(256) void pack_w(const float* __restrict__ w,
                                              _Float16* __restrict__ packW) {
    int d = blockIdx.x * 256 + threadIdx.x;           // dword id, 36864 total
    int j = d & 7;
    int frag = d >> 3;
    int lane = frag & 31;
    int tile = frag >> 5;
    int nt = tile & 7;
    int kt = tile >> 3;                                // 0..17
    int n = nt * 16 + (lane & 15);
    int k = kt * 32 + ((lane & 16) ? 8 : 0) + ((j < 4) ? 2 * j : 16 + 2 * (j - 4));
    packW[2 * d]     = (_Float16)w[(size_t)n * KTOT + k];
    packW[2 * d + 1] = (_Float16)w[(size_t)n * KTOT + k + 1];
}

__global__ __launch_bounds__(256) void pack_g(const float* __restrict__ g,
                                              _Float16* __restrict__ packG) {
    int d = blockIdx.x * 256 + threadIdx.x;           // 8192 dwords
    int j = d & 7;
    int frag = d >> 3;
    int lane = frag & 31;
    int tile = frag >> 5;
    int nt = tile & 7;
    int kt = tile >> 3;                                // 0..3
    int n = nt * 16 + (lane & 15);
    int k = kt * 32 + ((lane & 16) ? 8 : 0) + ((j < 4) ? 2 * j : 16 + 2 * (j - 4));
    packG[2 * d]     = (_Float16)g[(size_t)n * 128 + k];
    packG[2 * d + 1] = (_Float16)g[(size_t)n * 128 + k + 1];
}

// ---------------------------------------------------------------------------
// K2: partial conv as implicit GEMM (WMMA f16, fp32 accum).
// Tile: 32 pixels (contiguous in W) x 128 output channels per workgroup.
// 8 waves: wm in {0,1} (M0=wm*16), wn in 0..3 (32 N each wave via 2 frags).
// ---------------------------------------------------------------------------
#define ASTRIDE 584  // 576 + 8 halves pad; 1168B = 16B aligned per row

__global__ __launch_bounds__(256) void conv_gemm(const float* __restrict__ x,
                                                 const float* __restrict__ mask,
                                                 const _Float16* __restrict__ packW,
                                                 const float* __restrict__ bias,
                                                 const float* __restrict__ isig_p,
                                                 float* __restrict__ y,
                                                 float* __restrict__ mout) {
    __shared__ _Float16 Alds[32 * ASTRIDE];
    __shared__ float ratio_lds[32];

    const int blk = blockIdx.x;
    const int w0 = (blk & 7) * 32;
    const int h  = (blk >> 3) & 255;
    const int b  = blk >> 11;
    const int tid = threadIdx.x;

    // ---- stage im2col A tile (zero pad), K order = c*9 + (ky*3+kx) ----
    for (int idx = tid; idx < 32 * KTOT; idx += 256) {
        int m = idx & 31;
        int k = idx >> 5;
        int c = k / 9;
        int tap = k - c * 9;
        int ky = tap / 3, kx = tap - ky * 3;
        int hy = h + ky - 1;
        int wx = w0 + m + kx - 1;
        float v = 0.f;
        if ((unsigned)hy < (unsigned)H_ && (unsigned)wx < (unsigned)W_) {
            float mv = mask[((size_t)b * H_ + hy) * W_ + wx];
            v = x[(((size_t)b * CIN + c) * H_ + hy) * W_ + wx] * mv;
        }
        Alds[m * ASTRIDE + k] = (_Float16)v;
    }
    // ---- mask 3x3 sum -> ratio + valid-mask output ----
    if (tid < 32) {
        int w = w0 + tid;
        float ms = 0.f;
        for (int ky = 0; ky < 3; ++ky)
            for (int kx = 0; kx < 3; ++kx) {
                int hy = h + ky - 1, wx = w + kx - 1;
                if ((unsigned)hy < (unsigned)H_ && (unsigned)wx < (unsigned)W_)
                    ms += mask[((size_t)b * H_ + hy) * W_ + wx];
            }
        float ratio = (ms > 0.f) ? 9.f / fmaxf(ms, 1e-8f) : 0.f;
        ratio_lds[tid] = ratio;
        mout[((size_t)b * H_ + h) * W_ + w] = (ms > 0.f) ? 1.f : 0.f;
    }
    __syncthreads();

    const int wv = tid >> 5, lane = tid & 31;
    const int M0 = (wv & 1) * 16;
    const int wn = wv >> 1;                 // 0..3, N base = wn*64
    const int mrow = M0 + (lane & 15);
    const int koff = (lane < 16) ? 0 : 8;

    v8f acc[2];
#pragma unroll
    for (int f = 0; f < 2; ++f)
#pragma unroll
        for (int i = 0; i < 8; ++i) acc[f][i] = 0.f;

    for (int kt = 0; kt < 18; ++kt) {
        const int K0 = kt * 32;
        const v8h* Ap = reinterpret_cast<const v8h*>(&Alds[mrow * ASTRIDE + K0 + koff]);
        v8h lo = Ap[0];        // K0+koff   .. +7
        v8h hi = Ap[2];        // K0+koff+16 .. +23
        v16h a;
#pragma unroll
        for (int i = 0; i < 8; ++i) { a[i] = lo[i]; a[i + 8] = hi[i]; }

        if (kt + 1 < 18)
            __builtin_prefetch((const void*)&packW[(((size_t)(kt + 1) * 8 + wn * 2) * 32 + lane) * 16], 0, 1);

#pragma unroll
        for (int nf = 0; nf < 2; ++nf) {
            const int ntile = wn * 2 + nf;
            v16h bf = *reinterpret_cast<const v16h*>(&packW[(((size_t)kt * 8 + ntile) * 32 + lane) * 16]);
            acc[nf] = __builtin_amdgcn_wmma_f32_16x16x32_f16(false, a, false, bf,
                                                             (short)0, acc[nf], false, false);
        }
    }

    // ---- epilogue: out = valid ? acc/sigma * ratio + bias : 0 ----
    const float isg = isig_p[0];
    const int nlocal = lane & 15;
    const int mofs = (lane >= 16) ? 8 : 0;
#pragma unroll
    for (int nf = 0; nf < 2; ++nf) {
        const int n = wn * 64 + nf * 16 + nlocal;
        const float bs = bias[n];
        float* outp = &y[(((size_t)b * COUT + n) * H_ + h) * W_ + (w0 + M0 + mofs)];
#pragma unroll
        for (int r = 0; r < 8; ++r) {
            float ratio = ratio_lds[M0 + mofs + r];
            outp[r] = (ratio > 0.f) ? acc[nf][r] * isg * ratio + bs : 0.f;
        }
    }
}

// ---------------------------------------------------------------------------
// K3: partial GDN: z = y / sqrt(beta + gamma @ y^2) * mask  (WMMA GEMM K=128)
// Tile: 64 pixels x 128 channels; 8 waves: wm 0..3, wn 0..1 (4 frags each).
// ---------------------------------------------------------------------------
#define A2STRIDE 136  // 128 + 8 halves pad; 272B = 16B aligned per row

__global__ __launch_bounds__(256) void gdn_gemm(const float* __restrict__ y,
                                                const float* __restrict__ mbuf,
                                                const _Float16* __restrict__ packG,
                                                const float* __restrict__ beta,
                                                float* __restrict__ z) {
    __shared__ _Float16 A2[64 * A2STRIDE];
    __shared__ float mask64[64];

    const int blk = blockIdx.x;
    const int w0 = (blk & 3) * 64;
    const int h  = (blk >> 2) & 255;
    const int b  = blk >> 10;
    const int tid = threadIdx.x;

    for (int idx = tid; idx < 64 * 128; idx += 256) {
        int m = idx & 63;
        int c = idx >> 6;
        float v = y[(((size_t)b * COUT + c) * H_ + h) * W_ + (w0 + m)];
        A2[m * A2STRIDE + c] = (_Float16)(v * v);
    }
    if (tid < 64) mask64[tid] = mbuf[((size_t)b * H_ + h) * W_ + (w0 + tid)];
    __syncthreads();

    const int wv = tid >> 5, lane = tid & 31;
    const int M0 = (wv & 3) * 16;
    const int wn = wv >> 2;                 // 0..1, N base = wn*64
    const int mrow = M0 + (lane & 15);
    const int koff = (lane < 16) ? 0 : 8;

    v8f acc[4];
#pragma unroll
    for (int f = 0; f < 4; ++f)
#pragma unroll
        for (int i = 0; i < 8; ++i) acc[f][i] = 0.f;

#pragma unroll
    for (int kt = 0; kt < 4; ++kt) {
        const int K0 = kt * 32;
        const v8h* Ap = reinterpret_cast<const v8h*>(&A2[mrow * A2STRIDE + K0 + koff]);
        v8h lo = Ap[0];
        v8h hi = Ap[2];
        v16h a;
#pragma unroll
        for (int i = 0; i < 8; ++i) { a[i] = lo[i]; a[i + 8] = hi[i]; }
#pragma unroll
        for (int nf = 0; nf < 4; ++nf) {
            const int ntile = wn * 4 + nf;
            v16h bf = *reinterpret_cast<const v16h*>(&packG[(((size_t)kt * 8 + ntile) * 32 + lane) * 16]);
            acc[nf] = __builtin_amdgcn_wmma_f32_16x16x32_f16(false, a, false, bf,
                                                             (short)0, acc[nf], false, false);
        }
    }

    const int nlocal = lane & 15;
    const int mofs = (lane >= 16) ? 8 : 0;
#pragma unroll
    for (int nf = 0; nf < 4; ++nf) {
        const int n = wn * 64 + nf * 16 + nlocal;
        const float bt = beta[n];
        const size_t base = (((size_t)b * COUT + n) * H_ + h) * W_ + (w0 + M0 + mofs);
#pragma unroll
        for (int r = 0; r < 8; ++r) {
            float denom = sqrtf(bt + acc[nf][r]);
            float yv = y[base + r];
            z[base + r] = (yv / denom) * mask64[M0 + mofs + r];
        }
    }
}

// ---------------------------------------------------------------------------
// K4: mask row downsample (any-valid in reflect window, stride 2 along H)
// ---------------------------------------------------------------------------
__global__ __launch_bounds__(256) void mask_row(const float* __restrict__ m,
                                                float* __restrict__ mlo,
                                                float* __restrict__ mhi) {
    int gid = blockIdx.x * 256 + threadIdx.x;     // 8*128*256
    int w = gid & 255;
    int i = (gid >> 8) & 127;
    int b = gid >> 15;
    float a9 = 0.f, a7 = 0.f;
#pragma unroll
    for (int t = 0; t < 9; ++t) {
        int r = refl(2 * i + t - 4, H_);
        float v = m[((size_t)b * H_ + r) * W_ + w];
        a9 += v;
        if (t >= 1 && t <= 7) a7 += v;
    }
    size_t o = ((size_t)b * HH + i) * W_ + w;
    mlo[o] = (a9 > 0.f) ? 1.f : 0.f;
    mhi[o] = (a7 > 0.f) ? 1.f : 0.f;
}

// ---------------------------------------------------------------------------
// K5: DWT row pass (H filter, stride 2): z -> lo_r (h0), hi_r (h1)
// ---------------------------------------------------------------------------
__global__ __launch_bounds__(256) void dwt_rows(const float* __restrict__ z,
                                                float* __restrict__ rlo,
                                                float* __restrict__ rhi) {
    int gid = blockIdx.x * 256 + threadIdx.x;     // 1024 * 128 * 256
    int w = gid & 255;
    int i = (gid >> 8) & 127;
    int bc = gid >> 15;                            // 0..1023
    float s[9];
#pragma unroll
    for (int t = 0; t < 9; ++t) {
        int r = refl(2 * i + t - 4, H_);
        s[t] = z[((size_t)bc * H_ + r) * W_ + w];
    }
    float lo = 0.f, hi = 0.f;
#pragma unroll
    for (int t = 0; t < 9; ++t) lo += cH0[t] * s[t];
#pragma unroll
    for (int t = 0; t < 7; ++t) hi += cH1[t] * s[t + 1];
    size_t o = ((size_t)bc * HH + i) * W_ + w;
    rlo[o] = lo;
    rhi[o] = hi;
}

// ---------------------------------------------------------------------------
// K6: mask col downsample -> m_ll, m_lh, m_hl, m_hh (into d_out)
// ---------------------------------------------------------------------------
__global__ __launch_bounds__(256) void mask_col(const float* __restrict__ mlo,
                                                const float* __restrict__ mhi,
                                                float* __restrict__ out) {
    int gid = blockIdx.x * 256 + threadIdx.x;     // 8*128*128
    int j = gid & 127;
    int i = (gid >> 7) & 127;
    int b = gid >> 14;
    float l9 = 0.f, l7 = 0.f, h9 = 0.f, h7 = 0.f;
#pragma unroll
    for (int t = 0; t < 9; ++t) {
        int c = refl(2 * j + t - 4, W_);
        size_t idx = ((size_t)b * HH + i) * W_ + c;
        float vl = mlo[idx], vh = mhi[idx];
        l9 += vl; h9 += vh;
        if (t >= 1 && t <= 7) { l7 += vl; h7 += vh; }
    }
    size_t o = ((size_t)b * HH + i) * WH + j;
    out[OFF_MLL + o] = (l9 > 0.f) ? 1.f : 0.f;
    out[OFF_MLH + o] = (l7 > 0.f) ? 1.f : 0.f;
    out[OFF_MHL + o] = (h9 > 0.f) ? 1.f : 0.f;
    out[OFF_MHH + o] = (h7 > 0.f) ? 1.f : 0.f;
}

// ---------------------------------------------------------------------------
// K7: DWT col pass (W filter, stride 2): lo_r -> ll,lh ; hi_r -> hl,hh
// ---------------------------------------------------------------------------
__global__ __launch_bounds__(256) void dwt_cols(const float* __restrict__ rlo,
                                                const float* __restrict__ rhi,
                                                float* __restrict__ out) {
    int gid = blockIdx.x * 256 + threadIdx.x;     // 2 * 1024 * 128 * 128
    int j = gid & 127;
    int i = (gid >> 7) & 127;
    int bc = (gid >> 14) & 1023;
    int half = gid >> 24;                          // 0 = lo_r, 1 = hi_r
    const float* src = half ? rhi : rlo;
    float s[9];
#pragma unroll
    for (int t = 0; t < 9; ++t) {
        int c = refl(2 * j + t - 4, W_);
        s[t] = src[((size_t)bc * HH + i) * W_ + c];
    }
    float a = 0.f, bb = 0.f;
#pragma unroll
    for (int t = 0; t < 9; ++t) a += cH0[t] * s[t];
#pragma unroll
    for (int t = 0; t < 7; ++t) bb += cH1[t] * s[t + 1];
    size_t o = ((size_t)bc * HH + i) * WH + j;
    out[(half ? OFF_HL : OFF_LL) + o] = a;
    out[(half ? OFF_HH : OFF_LH) + o] = bb;
}

// ---------------------------------------------------------------------------
// kernel_launch
// ---------------------------------------------------------------------------
extern "C" void kernel_launch(void* const* d_in, const int* in_sizes, int n_in,
                              void* d_out, int out_size, void* d_ws, size_t ws_size,
                              hipStream_t stream) {
    const float* X     = (const float*)d_in[0];
    const float* Mk    = (const float*)d_in[1];
    const float* Wt    = (const float*)d_in[2];
    const float* Bias  = (const float*)d_in[3];
    const float* U     = (const float*)d_in[4];
    const float* Beta  = (const float*)d_in[5];
    const float* Gamma = (const float*)d_in[6];
    float* out = (float*)d_out;

    char* ws = (char*)d_ws;
    float*    sig   = (float*)(ws + 0);
    _Float16* packW = (_Float16*)(ws + (1 << 10));       // 147456 B
    _Float16* packG = (_Float16*)(ws + (1 << 18));       // 32768 B
    float*    mbuf  = (float*)(ws + 524288);             // 2 MB   [8,256,256]
    float*    mlo   = (float*)(ws + 3145728);            // 1 MB   [8,128,256]
    float*    mhi   = (float*)(ws + 4194304);            // 1 MB
    float*    ybuf  = (float*)(ws + 8388608);            // 256 MB [8,128,256,256]
    float*    zbuf  = (float*)(ws + 276824064);          // 256 MB
    float*    rlo   = ybuf;                              // reuse y region
    float*    rhi   = (float*)(ws + 8388608 + 134217728);

    sigma_kernel<<<1, 256, 0, stream>>>(Wt, U, sig);
    pack_w<<<144, 256, 0, stream>>>(Wt, packW);
    pack_g<<<32, 256, 0, stream>>>(Gamma, packG);
    conv_gemm<<<16384, 256, 0, stream>>>(X, Mk, packW, Bias, sig, ybuf, mbuf);
    gdn_gemm<<<8192, 256, 0, stream>>>(ybuf, mbuf, packG, Beta, zbuf);
    mask_row<<<1024, 256, 0, stream>>>(mbuf, mlo, mhi);
    dwt_rows<<<131072, 256, 0, stream>>>(zbuf, rlo, rhi);
    mask_col<<<512, 256, 0, stream>>>(mlo, mhi, out);
    dwt_cols<<<131072, 256, 0, stream>>>(rlo, rhi, out);
}
